// self_attention_47467978556181
// MI455X (gfx1250) — matrix operand order repaired and checked by reading, not compiled, not run on hardware
//
#include <hip/hip_runtime.h>
#include <stdint.h>

// ---------------------------------------------------------------------------
// Self-attention (B=4, S=2048, D=1024) for gfx1250 using v_wmma_f32_16x16x32_bf16
// ---------------------------------------------------------------------------

typedef __attribute__((ext_vector_type(16))) __bf16 v16bf;
typedef __attribute__((ext_vector_type(8)))  float  v8f;

#define BATCH 4
#define SDIM  2048
#define DDIM  1024

__device__ __forceinline__ unsigned short f2bf(float x) {
  union { float f; uint32_t u; } c; c.f = x;
  uint32_t r = c.u + 0x7FFFu + ((c.u >> 16) & 1u);   // round-to-nearest-even
  return (unsigned short)(r >> 16);
}

union FragU { uint4 q[2]; v16bf v; };

// A fragment: 16x32 bf16, row-major A, lanes 0-15 = rows 0-15 holding K 0-7,16-23;
// lanes 16-31 = rows 0-15 holding K 8-15,24-31. (CDNA5 ISA 7.12.2)
__device__ __forceinline__ v16bf load_frag_a(const unsigned short* __restrict__ A,
                                             int lda, int row0, int k0, int lane) {
  int m   = lane & 15;
  int sel = (lane & 16) ? 8 : 0;
  const unsigned short* p = A + (size_t)(row0 + m) * lda + (k0 + sel);
  FragU f;
  f.q[0] = *(const uint4*)(p);        // K sel .. sel+7
  f.q[1] = *(const uint4*)(p + 16);   // K sel+16 .. sel+23
  return f.v;
}

// B fragment: 32x16 bf16. Memory operand is N-major (N rows of K values =
// "B transposed"), so each lane (column n) reads two runs of 8 K values:
// lanes 0-15 -> K 0-7,16-23, lanes 16-31 -> K 8-15,24-31.
__device__ __forceinline__ v16bf load_frag_b(const unsigned short* __restrict__ Bm,
                                             int ldb, int col0, int k0, int lane) {
  int n   = lane & 15;
  int sel = (lane & 16) ? 8 : 0;
  const unsigned short* p = Bm + (size_t)(col0 + n) * ldb + (k0 + sel);
  FragU f;
  f.q[0] = *(const uint4*)(p);
  f.q[1] = *(const uint4*)(p + 16);
  return f.v;
}

// ---------------------------------------------------------------------------
// Generic bf16 GEMM:  C[M,N] = A[M,K] * B[N,K]^T  (A,B bf16; accumulate f32)
// Block: 256 threads = 8 waves (2 in M x 4 in N); wave tile 64x64;
// block tile 128(M) x 256(N).  MODE: 0 = f32 row-major out,
// 1 = bf16 row-major out, 2 = bf16 transposed-per-batch out (V^T: [b][n][s]).
// ---------------------------------------------------------------------------
template <int MODE>
__global__ __launch_bounds__(256)
void gemm_bf16(const unsigned short* __restrict__ A,
               const unsigned short* __restrict__ Bm,
               void* __restrict__ C,
               int K, int lda, int ldb, int ldc) {
  const int lane = threadIdx.x & 31;
  const int wave = threadIdx.x >> 5;
  const int m0 = blockIdx.y * 128 + (wave >> 2) * 64;
  const int n0 = blockIdx.x * 256 + (wave & 3) * 64;

  v8f acc[4][4];
#pragma unroll
  for (int i = 0; i < 4; ++i)
#pragma unroll
    for (int j = 0; j < 4; ++j) acc[i][j] = (v8f){0.f,0.f,0.f,0.f,0.f,0.f,0.f,0.f};

  for (int k0 = 0; k0 < K; k0 += 32) {
    v16bf af[4], bf[4];
#pragma unroll
    for (int i = 0; i < 4; ++i) af[i] = load_frag_a(A, lda, m0 + 16 * i, k0, lane);
#pragma unroll
    for (int j = 0; j < 4; ++j) bf[j] = load_frag_b(Bm, ldb, n0 + 16 * j, k0, lane);
#pragma unroll
    for (int i = 0; i < 4; ++i)
#pragma unroll
      for (int j = 0; j < 4; ++j)
        acc[i][j] = __builtin_amdgcn_wmma_f32_16x16x32_bf16(
            false, af[i], false, bf[j], (short)0, acc[i][j], false, false);
  }

  // C layout: VGPR r, lanes 0-15 -> row r, lanes 16-31 -> row r+8; col = lane%16
  const int nn = lane & 15;
  const int mb = (lane & 16) ? 8 : 0;
#pragma unroll
  for (int i = 0; i < 4; ++i) {
#pragma unroll
    for (int j = 0; j < 4; ++j) {
#pragma unroll
      for (int r = 0; r < 8; ++r) {
        const int gm = m0 + 16 * i + mb + r;
        const int gn = n0 + 16 * j + nn;
        const float val = acc[i][j][r];
        if (MODE == 0) {
          ((float*)C)[(size_t)gm * ldc + gn] = val;
        } else if (MODE == 1) {
          ((unsigned short*)C)[(size_t)gm * ldc + gn] = f2bf(val);
        } else {  // MODE == 2: V^T store, gm = b*SDIM + s, out[b][gn][s]
          const int b = gm >> 11;           // SDIM = 2048
          const int s = gm & (SDIM - 1);
          ((unsigned short*)C)[((size_t)b * DDIM + gn) * SDIM + s] = f2bf(val);
        }
      }
    }
  }
}

// ---------------------------------------------------------------------------
// f32 -> bf16 bulk convert (4 elements per thread, n divisible by 1024)
// ---------------------------------------------------------------------------
__global__ __launch_bounds__(256)
void cvt_f32_bf16(const float* __restrict__ src, unsigned short* __restrict__ dst, int n) {
  int i = (blockIdx.x * 256 + threadIdx.x) * 4;
  if (i >= n) return;
  float4 f = *(const float4*)(src + i);
  ushort4 o;
  o.x = f2bf(f.x); o.y = f2bf(f.y); o.z = f2bf(f.z); o.w = f2bf(f.w);
  *(ushort4*)(dst + i) = o;
}

// ---------------------------------------------------------------------------
// Row softmax over n=2048 columns; one 256-thread block per row.
// Reads f32 scores, writes bf16 probabilities.
// ---------------------------------------------------------------------------
__global__ __launch_bounds__(256)
void softmax_rows(const float* __restrict__ S, unsigned short* __restrict__ P, int n) {
  __shared__ float red[256];
  const int tid = threadIdx.x;
  const size_t row = blockIdx.x;
  const float* sr = S + row * (size_t)n;

  float v[8];
  float mx = -3.402823466e38f;
#pragma unroll
  for (int j = 0; j < 8; ++j) {
    v[j] = sr[tid + j * 256];
    mx = fmaxf(mx, v[j]);
  }
  red[tid] = mx; __syncthreads();
  for (int s = 128; s > 0; s >>= 1) {
    if (tid < s) red[tid] = fmaxf(red[tid], red[tid + s]);
    __syncthreads();
  }
  const float rowmax = red[0];
  __syncthreads();

  float sum = 0.f;
#pragma unroll
  for (int j = 0; j < 8; ++j) {
    v[j] = __expf(v[j] - rowmax);
    sum += v[j];
  }
  red[tid] = sum; __syncthreads();
  for (int s = 128; s > 0; s >>= 1) {
    if (tid < s) red[tid] += red[tid + s];
    __syncthreads();
  }
  const float inv = 1.0f / red[0];

  unsigned short* pr = P + row * (size_t)n;
#pragma unroll
  for (int j = 0; j < 8; ++j) pr[tid + j * 256] = f2bf(v[j] * inv);
}

// ---------------------------------------------------------------------------
// Host orchestration
// ---------------------------------------------------------------------------
extern "C" void kernel_launch(void* const* d_in, const int* in_sizes, int n_in,
                              void* d_out, int out_size, void* d_ws, size_t ws_size,
                              hipStream_t stream) {
  const float* x  = (const float*)d_in[0];   // [B,S,D]
  const float* wq = (const float*)d_in[1];   // [D,D]
  const float* wk = (const float*)d_in[2];
  const float* wv = (const float*)d_in[3];
  float* out = (float*)d_out;                // [B,S,D]

  char* ws = (char*)d_ws;
  const size_t MB = 1u << 20;
  unsigned short* wq_b = (unsigned short*)(ws + 0 * MB);    // 2 MB
  unsigned short* wk_b = (unsigned short*)(ws + 2 * MB);    // 2 MB
  unsigned short* wv_b = (unsigned short*)(ws + 4 * MB);    // 2 MB
  unsigned short* x_b  = (unsigned short*)(ws + 6 * MB);    // 16 MB
  unsigned short* q_b  = (unsigned short*)(ws + 22 * MB);   // 16 MB
  unsigned short* k_b  = (unsigned short*)(ws + 38 * MB);   // 16 MB
  unsigned short* vT_b = (unsigned short*)(ws + 54 * MB);   // 16 MB  [b][d][s]
  float*          sc   = (float*)         (ws + 70 * MB);   // 16 MB  per-batch scores
  unsigned short* p_b  = (unsigned short*)(ws + 86 * MB);   // 8 MB   per-batch probs

  const int BS = BATCH * SDIM;               // 8192
  const int DD = DDIM * DDIM;                // 1 M
  const int XN = BS * DDIM;                  // 8 M

  // 1) convert weights and activations to bf16
  cvt_f32_bf16<<<DD / 1024, 256, 0, stream>>>(wq, wq_b, DD);
  cvt_f32_bf16<<<DD / 1024, 256, 0, stream>>>(wk, wk_b, DD);
  cvt_f32_bf16<<<DD / 1024, 256, 0, stream>>>(wv, wv_b, DD);
  cvt_f32_bf16<<<XN / 1024, 256, 0, stream>>>(x,  x_b,  XN);

  // 2) projections: Q = x W_q^T, K = x W_k^T (bf16 row-major), V^T (bf16 [b][d][s])
  // M = 8192, N = 1024, K = 1024 -> grid (N/256, M/128)
  dim3 gproj(DDIM / 256, BS / 128);
  gemm_bf16<1><<<gproj, 256, 0, stream>>>(x_b, wq_b, q_b,  DDIM, DDIM, DDIM, DDIM);
  gemm_bf16<1><<<gproj, 256, 0, stream>>>(x_b, wk_b, k_b,  DDIM, DDIM, DDIM, DDIM);
  gemm_bf16<2><<<gproj, 256, 0, stream>>>(x_b, wv_b, vT_b, DDIM, DDIM, DDIM, 0);

  // 3) per-batch attention (scores buffer reused sequentially on the stream)
  dim3 gsc(SDIM / 256, SDIM / 128);   // M=2048, N=2048
  dim3 gpv(DDIM / 256, SDIM / 128);   // M=2048, N=1024
  for (int b = 0; b < BATCH; ++b) {
    const unsigned short* qb = q_b  + (size_t)b * SDIM * DDIM;
    const unsigned short* kb = k_b  + (size_t)b * SDIM * DDIM;
    const unsigned short* vb = vT_b + (size_t)b * DDIM * SDIM;
    float* ob = out + (size_t)b * SDIM * DDIM;

    // scores = Q K^T (f32), K row-major is exactly the N-major B operand
    gemm_bf16<0><<<gsc, 256, 0, stream>>>(qb, kb, sc, DDIM, DDIM, DDIM, SDIM);
    // row softmax -> bf16 P
    softmax_rows<<<SDIM, 256, 0, stream>>>(sc, p_b, SDIM);
    // out = P V ; V^T ([d][s]) is the N-major B operand, K = S = 2048
    gemm_bf16<0><<<gpv, 256, 0, stream>>>(p_b, vb, ob, SDIM, SDIM, SDIM, DDIM);
  }
}
